// AffEncoder_15917148799531
// MI455X (gfx1250) — matrix operand order repaired and checked by reading, not compiled, not run on hardware
//
#include <hip/hip_runtime.h>

// ---- CDNA5 (gfx1250) types ----
typedef _Float16 v16h __attribute__((ext_vector_type(16)));
typedef float    v8f  __attribute__((ext_vector_type(8)));
typedef unsigned int u32x4 __attribute__((ext_vector_type(4)));
typedef int          i32x4 __attribute__((ext_vector_type(4)));
typedef int          i32x8 __attribute__((ext_vector_type(8)));

union HF { v16h h; u32x4 u[2]; };   // 32B fragment: 16 halves <-> 2 dwordx4

#define N_   256
#define T_   1024
#define TT   64
#define LEAKF 0.01f

// LDS arena (bytes), phase-overlapped:
//  [0,      23040)  F1  f16 [80][3][48]   feat1 (stage1 out / stage2 B source)
//                   phase C: WC1 f32[16][48][5] (15360) + WC2 f32[8][16][3]
//  [23040,  36864)  F3  f32 [48][72]      feat3_in (stage2 out / phase C in)
//                   phase 0: XS f32[88][27] TDM staging (9504 B)
//  [36864, 119808)  WF  f16 [9][16][288]  folded stage1 weights (82944 B)
//                   stage 2: W2H f16 [48][448] (43008 B)
//  [119808,125440)  XH  f16 [88][32]      stage1 B operand, k-contiguous
//                   phase C: H f32[16][68] (4352 B)
#define OFF_F3  23040
#define OFF_WF  36864
#define OFF_XH  119808
#define ARENA_B 125440

__global__ __launch_bounds__(256, 2)
void affenc_fused(const float* __restrict__ poses,
                  const float* __restrict__ A1,  const float* __restrict__ A2,
                  const float* __restrict__ W1,  const float* __restrict__ b1,
                  const float* __restrict__ W2,  const float* __restrict__ b2,
                  const float* __restrict__ Wc1, const float* __restrict__ bc1,
                  const float* __restrict__ bn1w, const float* __restrict__ bn1b,
                  const float* __restrict__ bn1m, const float* __restrict__ bn1v,
                  const float* __restrict__ Wc2, const float* __restrict__ bc2,
                  const float* __restrict__ bn2w, const float* __restrict__ bn2b,
                  const float* __restrict__ bn2m, const float* __restrict__ bn2v,
                  float* __restrict__ out)
{
    const int tid  = threadIdx.x;
    const int lane = tid & 31;
    const int wid  = tid >> 5;
    const int tile = blockIdx.x;          // 0..15
    const int n    = blockIdx.y;          // 0..255
    const int t0   = tile * TT;

    __shared__ __attribute__((aligned(16))) unsigned char arena[ARENA_B];
    _Float16* F1  = (_Float16*)(arena);              // [80][3][48], time base t0-7
    float*    F3  = (float*)(arena + OFF_F3);        // [48][72],    time base t0-3
    float*    XS  = (float*)(arena + OFF_F3);        // [88][27],    time base t0-11 (early)
    _Float16* WF  = (_Float16*)(arena + OFF_WF);     // [9][16][288]
    _Float16* W2H = (_Float16*)(arena + OFF_WF);     // [48][448]    (stage 2)
    _Float16* XH  = (_Float16*)(arena + OFF_XH);     // [88][32]

    __shared__ float s_a1[405];                      // A1 [5][9][9]
    __shared__ float s_a2[27];                       // A2 [3][3][3]
    __shared__ float s_b1[16 * 9];                   // folded bias1[c][w]
    __shared__ float s_sc1[16], s_sh1[16], s_sc2[8], s_sh2[8];

    // ---------------- Phase 0: stage pose tile (TDM) + small constants -------------
    long long row0 = (long long)n * T_ + t0 - 11;
    if (row0 >= 0) {
        if (wid == 0) {
            unsigned long long ga =
                (unsigned long long)(const void*)poses + (unsigned long long)row0 * 27ull * 4ull;
            unsigned ldsoff = (unsigned)(unsigned long long)(void*)XS;
            u32x4 g0 = (u32x4)0u;
            g0[0] = 1u;                                   // count=1
            g0[1] = ldsoff;                               // lds_addr
            g0[2] = (unsigned)(ga & 0xffffffffull);
            g0[3] = (unsigned)((ga >> 32) & 0x01ffffffull) | (2u << 30); // type=2
            i32x8 g1 = (i32x8)0;
            g1[0] = (int)(2u << 16);                      // data_size = 4B
            g1[1] = (int)(27u << 16);                     // tensor_dim0 = 27
            g1[2] = 0;
            g1[3] = (int)((262144u >> 16) | (27u << 16)); // dim1_hi16 | tile_dim0=27
            g1[4] = 88;                                   // tile_dim1 = 88 rows
            g1[5] = 27;                                   // tensor_dim0_stride
            g1[6] = 0; g1[7] = 0;
            i32x4 gz = (i32x4)0;
#if __clang_major__ >= 23
            i32x8 gz8 = (i32x8)0;
            __builtin_amdgcn_tensor_load_to_lds(g0, g1, gz, gz, gz8, 0);
#else
            __builtin_amdgcn_tensor_load_to_lds(g0, g1, gz, gz, 0);
#endif
            __builtin_amdgcn_s_wait_tensorcnt(0);
        }
    } else {
        for (int i = tid; i < 88 * 27; i += 256) {
            long long rr = row0 + (i / 27);
            XS[i] = (rr >= 0) ? poses[rr * 27 + (i % 27)] : 0.f;
        }
    }
    for (int i = tid; i < 405; i += 256) s_a1[i] = A1[i];
    if (tid < 27) s_a2[tid] = A2[tid];
    if (tid < 16) { float sc = bn1w[tid] * rsqrtf(bn1v[tid] + 1e-5f);
                    s_sc1[tid] = sc;
                    s_sh1[tid] = bn1b[tid] - bn1m[tid] * sc + bc1[tid] * sc; }
    if (tid < 8)  { float sc = bn2w[tid] * rsqrtf(bn2v[tid] + 1e-5f);
                    s_sc2[tid] = sc;
                    s_sh2[tid] = bn2b[tid] - bn2m[tid] * sc + bc2[tid] * sc; }
    __syncthreads();

    // --------- Phase 1: build fragment-layout operands (once, cooperatively) -------
    // XH[t][cv]: cv = ci*9+v (27 real, pad 32) so every 16-aligned k-block of
    // k = dt*32+cv is 32 contiguous bytes -> B-frag = 2x ds_load_b128.
    for (int i = tid; i < 88 * 32; i += 256) {
        int t = i >> 5, cv = i & 31;
        float val = 0.f;
        if (cv < 27) { int ci = cv / 9, v = cv - ci * 9; val = XS[t * 27 + v * 3 + ci]; }
        XH[i] = (_Float16)val;
    }
    // WF[w][c][k]: folded conv1 x einsum1 weights, Wfold = sum_k1 W1 * A1.
    for (int i = tid; i < 9 * 16 * 288; i += 256) {
        int w = i / 4608, r = i - w * 4608;
        int c = r / 288,  k = r - c * 288;
        int dt = k >> 5, cv = k & 31;
        float val = 0.f;
        if (cv < 27) {
            int ci = cv / 9, v = cv - ci * 9;
            #pragma unroll
            for (int k1 = 0; k1 < 5; ++k1)
                val += W1[(k1 * 16 + c) * 27 + ci * 9 + dt] * s_a1[(k1 * 9 + v) * 9 + w];
        }
        WF[i] = (_Float16)val;
    }
    if (tid < 144) {   // bias1[c][w] = sum_{k1,v} b1[k1*16+c] * A1[k1,v,w]
        int c = tid / 9, w = tid % 9;
        float s = 0.f;
        for (int k1 = 0; k1 < 5; ++k1) {
            float bb = b1[k1 * 16 + c];
            for (int v = 0; v < 9; ++v) s += bb * s_a1[(k1 * 9 + v) * 9 + w];
        }
        s_b1[tid] = s;
    }
    __syncthreads();   // XS dead; WF/XH ready

    for (int i = tid; i < 48 * 72; i += 256) F3[i] = 0.f;          // zero accum
    for (int i = tid; i < 2 * 144; i += 256) F1[78 * 144 + i] = (_Float16)0.f; // pad rows

    // WMMA lane decomposition (wave32, 16x16x32 f16)
    const int m   = lane & 15;
    const int kbA = (lane & 16) ? 16 : 0;
    const int kbB = (lane & 16) ? 16 : 0;
    const int mb  = (lane & 16) ? 8  : 0;

    // ---------------- Stage 1: folded conv1+einsum1 -> F1 (f16) --------------------
    for (int w = wid; w < 9; w += 8) {
        const int p = w / 3, e = w % 3;
        v8f acc[5] = {};
        const _Float16* wfrow = WF + (w * 16 + m) * 288;
        for (int ks = 0; ks < 9; ++ks) {
            HF a;
            const u32x4* ap = (const u32x4*)(wfrow + ks * 32 + kbA);
            a.u[0] = ap[0];                 // halves kbA .. kbA+7
            a.u[1] = ap[1];                 // halves kbA+8 .. kbA+15
            #pragma unroll
            for (int nt = 0; nt < 5; ++nt) {
                int tl = nt * 16 + m;       // output local time (base t0-7)
                HF b;
                const u32x4* bp = (const u32x4*)(XH + (tl + ks) * 32 + kbB);
                b.u[0] = bp[0];
                b.u[1] = bp[1];
                acc[nt] = __builtin_amdgcn_wmma_f32_16x16x32_f16(
                              false, a.h, false, b.h, (short)0, acc[nt], false, false);
            }
        }
        #pragma unroll
        for (int nt = 0; nt < 5; ++nt) {
            int tl = nt * 16 + m;
            if (tl < 78) {
                #pragma unroll
                for (int r = 0; r < 8; ++r) {
                    int c = mb + r;
                    F1[(tl * 3 + p) * 48 + (c * 3 + e)] =
                        (_Float16)(acc[nt][r] + s_b1[c * 9 + w]);
                }
            }
        }
    }
    __syncthreads();

    // Convert W2 -> f16 [48][448] over the dead WF region (zero-padded past 432).
    for (int i = tid; i < 48 * 448; i += 256) {
        int mr = i / 448, k = i - mr * 448;
        float val = 0.f;
        if (k < 432) { int dt = k / 48, ce = k - dt * 48; val = W2[mr * 432 + ce * 9 + dt]; }
        W2H[i] = (_Float16)val;
    }
    __syncthreads();

    // ---------------- Stage 2: conv2 GEMM + fused einsum2 -> F3 --------------------
    // k = dt*48+ce; 48 = 3*16 so 16-aligned k-blocks are contiguous in F1[t][p][ce].
    for (int f = wid; f < 45; f += 8) {
        const int p  = f / 15;
        const int k2 = (f % 15) / 5;
        const int nt = f % 5;
        const int tly = nt * 16 + m;            // y2 local time (base t0-3)
        v8f acc = {};
        const _Float16* w2row = W2H + (k2 * 16 + m) * 448;
        for (int ks = 0; ks < 14; ++ks) {
            HF a, b;
            const u32x4* ap = (const u32x4*)(w2row + ks * 32 + kbA);
            a.u[0] = ap[0];
            a.u[1] = ap[1];
            int k0 = ks * 32 + kbB;
            if (k0 < 432) {
                int dt = k0 / 48, ce0 = k0 - dt * 48;    // ce0 in {0,16,32}
                int tx = tly + dt; if (tx > 79) tx = 79;
                const u32x4* bp = (const u32x4*)(F1 + (tx * 3 + p) * 48 + ce0);
                b.u[0] = bp[0];
                b.u[1] = bp[1];
            } else {
                b.u[0] = (u32x4)0u; b.u[1] = (u32x4)0u;
            }
            acc = __builtin_amdgcn_wmma_f32_16x16x32_f16(
                      false, a.h, false, b.h, (short)0, acc, false, false);
        }
        if (tly < 70) {   // einsum2 fused at D-store
            #pragma unroll
            for (int r = 0; r < 8; ++r) {
                int c2 = mb + r;
                float dv = acc[r] + b2[k2 * 16 + c2];
                #pragma unroll
                for (int w2 = 0; w2 < 3; ++w2)
                    atomicAdd(&F3[(c2 * 3 + w2) * 72 + tly],
                              dv * s_a2[(k2 * 3 + p) * 3 + w2]);
            }
        }
    }
    __syncthreads();

    // ---------------- Phase C: temporal 1-D convs + BN + leaky ---------------------
    float* WC1 = (float*)(arena);                  // [16][48][5]  (over dead F1)
    float* WC2 = (float*)(arena + 15360);          // [8][16][3]
    float* H   = (float*)(arena + OFF_XH);         // [16][68]     (over dead XH)
    for (int i = tid; i < 16 * 48 * 5; i += 256) WC1[i] = Wc1[i];
    for (int i = tid; i < 8 * 16 * 3;  i += 256) WC2[i] = Wc2[i];
    __syncthreads();

    for (int idx = tid; idx < 16 * 66; idx += 256) {   // h, time base t0-1
        int oc = idx / 66, tl = idx % 66;
        float s = 0.f;
        for (int ic = 0; ic < 48; ++ic) {
            const float* wr = &WC1[(oc * 48 + ic) * 5];
            const float* fr = &F3[ic * 72 + tl];
            #pragma unroll
            for (int kt = 0; kt < 5; ++kt) s += fr[kt] * wr[kt];
        }
        float z = s * s_sc1[oc] + s_sh1[oc];
        H[oc * 68 + tl] = (z > 0.f) ? z : LEAKF * z;
    }
    __syncthreads();

    for (int idx = tid; idx < 8 * 64; idx += 256) {    // final output, time base t0
        int oc = idx / 64, tl = idx % 64;
        float s = 0.f;
        #pragma unroll
        for (int ic = 0; ic < 16; ++ic) {
            const float* wr = &WC2[(oc * 16 + ic) * 3];
            const float* hr = &H[ic * 68 + tl];
            s += hr[0] * wr[0] + hr[1] * wr[1] + hr[2] * wr[2];
        }
        float z = s * s_sc2[oc] + s_sh2[oc];
        z = (z > 0.f) ? z : LEAKF * z;
        out[((long long)n * T_ + (t0 + tl)) * 8 + oc] = z;
    }
}

extern "C" void kernel_launch(void* const* d_in, const int* in_sizes, int n_in,
                              void* d_out, int out_size, void* d_ws, size_t ws_size,
                              hipStream_t stream) {
    (void)in_sizes; (void)n_in; (void)out_size; (void)d_ws; (void)ws_size;
    const float* poses = (const float*)d_in[0];
    const float* A1    = (const float*)d_in[1];
    const float* A2    = (const float*)d_in[2];
    const float* W1    = (const float*)d_in[3];
    const float* b1    = (const float*)d_in[4];
    const float* W2    = (const float*)d_in[5];
    const float* b2    = (const float*)d_in[6];
    const float* Wc1   = (const float*)d_in[7];
    const float* bc1   = (const float*)d_in[8];
    const float* bn1w  = (const float*)d_in[9];
    const float* bn1b  = (const float*)d_in[10];
    const float* bn1m  = (const float*)d_in[11];
    const float* bn1v  = (const float*)d_in[12];
    const float* Wc2   = (const float*)d_in[13];
    const float* bc2   = (const float*)d_in[14];
    const float* bn2w  = (const float*)d_in[15];
    const float* bn2b  = (const float*)d_in[16];
    const float* bn2m  = (const float*)d_in[17];
    const float* bn2v  = (const float*)d_in[18];

    dim3 grid(T_ / TT, N_);
    affenc_fused<<<grid, 256, 0, stream>>>(poses, A1, A2, W1, b1, W2, b2, Wc1, bc1,
                                           bn1w, bn1b, bn1m, bn1v, Wc2, bc2,
                                           bn2w, bn2b, bn2m, bn2v, (float*)d_out);
}
